// MySelfAttention_74096775790730
// MI455X (gfx1250) — matrix operand order repaired and checked by reading, not compiled, not run on hardware
//
#include <hip/hip_runtime.h>

#define D_DIM 1024
#define S_DIM 2048
#define B_DIM 4
#define R_TOT (B_DIM * S_DIM) /* 8192 total rows of x/Q/K/V */

#define TLD 40                 /* padded LDS row stride (elements): 80B = 20 dwords -> no bank conflicts */
#define TILE_ELEMS (64 * TLD)  /* one 64x32 tile (padded) */

typedef __bf16 bf16;
typedef __attribute__((ext_vector_type(16))) __bf16 v16bf;
typedef __attribute__((ext_vector_type(8)))  __bf16 v8bf;
typedef __attribute__((ext_vector_type(8)))  float  v8f;
typedef __attribute__((ext_vector_type(4)))  unsigned int v4u;
typedef int v4i_gcc __attribute__((vector_size(16)));  /* matches builtin param type */

#if defined(__AMDGCN__) && __has_builtin(__builtin_amdgcn_global_load_async_to_lds_b128) && \
    __has_builtin(__builtin_amdgcn_s_wait_asynccnt)
#define HAVE_ASYNC 1
#else
#define HAVE_ASYNC 0
#endif

static __device__ __forceinline__ v8f wmma_bf16(v16bf a, v16bf b, v8f c) {
  return __builtin_amdgcn_wmma_f32_16x16x32_bf16(
      /*neg_a=*/false, a, /*neg_b=*/false, b,
      /*c_mod=*/(short)0, c, /*reuse_a=*/false, /*reuse_b=*/false);
}

// One 16-byte global -> LDS copy, async (ASYNCcnt) when available.
static __device__ __forceinline__ void copy16(const bf16* g, bf16* l) {
#if HAVE_ASYNC
  __builtin_amdgcn_global_load_async_to_lds_b128(
      (__attribute__((address_space(1))) v4i_gcc*)g,
      (__attribute__((address_space(3))) v4i_gcc*)l,
      /*offset=*/0, /*cpol=*/0);
#else
  *(v4u*)l = *(const v4u*)g;
#endif
}

template <int N>
static __device__ __forceinline__ void async_wait() {
#if HAVE_ASYNC
  __builtin_amdgcn_s_wait_asynccnt(N);
#endif
}

// A fragment 16x32 bf16 from LDS tile (row-major, TLD stride).
// Lane L: M = L%16; VGPR0-3: K = off+[0..7]; VGPR4-7: K = 16+off+[0..7], off = (L<16?0:8).
static __device__ __forceinline__ v16bf lds_a_frag(const bf16* t, int row_local, int lhalf) {
  const v8bf* p = (const v8bf*)(t + row_local * TLD + lhalf * 8);
  union { v16bf v; v8bf h[2]; } u;
  u.h[0] = p[0];
  u.h[1] = p[2];  // +16 elements
  return u.v;
}

// B fragment 32x16 where B[k][n] = tile[col_local+n][k] (K contiguous per "row").
// Lane L: N = L%16; VGPR0-3: K = off+[0..7]; VGPR4-7: K = 16+off+[0..7], off = (L<16?0:8).
static __device__ __forceinline__ v16bf lds_b_frag(const bf16* t, int col_local, int lhalf) {
  const v8bf* p = (const v8bf*)(t + col_local * TLD + lhalf * 8);
  union { v16bf v; v8bf h[2]; } u;
  u.h[0] = p[0];
  u.h[1] = p[2];
  return u.v;
}

static __device__ __forceinline__ float redmax16(float v) {
#pragma unroll
  for (int m = 1; m < 16; m <<= 1) v = fmaxf(v, __shfl_xor(v, m, 32));
  return v;
}
static __device__ __forceinline__ float redsum16(float v) {
#pragma unroll
  for (int m = 1; m < 16; m <<= 1) v += __shfl_xor(v, m, 32);
  return v;
}

// -----------------------------------------------------------------------------
// Double-buffered LDS-tiled 64x64 GEMM core (K step 32), 128 threads / 4 waves.
// Ag: 64 rows (K contiguous, stride lda). Bg: 64 "columns" stored as K-contiguous
// rows (stride ldb). Per k-step each thread async-copies 4x16B (2 for A, 2 for B);
// ASYNCcnt per wave per tile = 4, so waiting <=4 after issuing the next tile
// guarantees the current tile landed (async loads retire in order).
// -----------------------------------------------------------------------------
template <int NK>
static __device__ __forceinline__ void gemm_tile(
    const bf16* __restrict__ Ag, int lda,
    const bf16* __restrict__ Bg, int ldb,
    bf16* sA, bf16* sB,  /* each 2*TILE_ELEMS */
    v8f acc[4], int tid, int wave, int lhalf, int lmod) {
  const int srow  = tid >> 1;         // 0..63
  const int skoff = (tid & 1) * 16;   // 0 or 16

  // Prologue: stage tile 0 into buffer 0.
  copy16(Ag + (size_t)srow * lda + skoff,     sA + srow * TLD + skoff);
  copy16(Ag + (size_t)srow * lda + skoff + 8, sA + srow * TLD + skoff + 8);
  copy16(Bg + (size_t)srow * ldb + skoff,     sB + srow * TLD + skoff);
  copy16(Bg + (size_t)srow * ldb + skoff + 8, sB + srow * TLD + skoff + 8);

  for (int k0 = 0; k0 < NK; k0 += 32) {
    const int buf = (k0 >> 5) & 1;
    if (k0 + 32 < NK) {
      const int nb = buf ^ 1;
      const int kn = k0 + 32;
      copy16(Ag + (size_t)srow * lda + kn + skoff,     sA + nb * TILE_ELEMS + srow * TLD + skoff);
      copy16(Ag + (size_t)srow * lda + kn + skoff + 8, sA + nb * TILE_ELEMS + srow * TLD + skoff + 8);
      copy16(Bg + (size_t)srow * ldb + kn + skoff,     sB + nb * TILE_ELEMS + srow * TLD + skoff);
      copy16(Bg + (size_t)srow * ldb + kn + skoff + 8, sB + nb * TILE_ELEMS + srow * TLD + skoff + 8);
      async_wait<4>();   // retire current tile's 4 ops; next tile's 4 stay in flight
    } else {
      async_wait<0>();
    }
    __syncthreads();     // all waves' portions of the current tile visible

    const bf16* tA = sA + buf * TILE_ELEMS;
    const bf16* tB = sB + buf * TILE_ELEMS;
    v16bf a = lds_a_frag(tA, wave * 16 + lmod, lhalf);
#pragma unroll
    for (int n = 0; n < 4; n++) {
      v16bf b = lds_b_frag(tB, n * 16 + lmod, lhalf);
      acc[n] = wmma_bf16(a, b, acc[n]);
    }
    __syncthreads();     // done reading buf before it is overwritten
  }
}

__global__ void cvt_f32_bf16(const float* __restrict__ s, bf16* __restrict__ d, int n) {
  int i = blockIdx.x * blockDim.x + threadIdx.x;
  if (i < n) d[i] = (bf16)s[i];
}

// -----------------------------------------------------------------------------
// QKV projection: out[r,e] = sum_d xb[r,d] * W[e,d] + b[e]
// z = 0: Q (scaled by 1/sqrt(D)=1/32), z = 1: K, z = 2: V stored transposed.
// -----------------------------------------------------------------------------
__global__ __launch_bounds__(128) void qkv_gemm(
    const bf16* __restrict__ xb,
    const bf16* __restrict__ wq, const bf16* __restrict__ wk, const bf16* __restrict__ wv,
    const float* __restrict__ bq, const float* __restrict__ bk, const float* __restrict__ bv,
    bf16* __restrict__ qo, bf16* __restrict__ ko, bf16* __restrict__ vt) {
  __shared__ bf16 sA[2 * TILE_ELEMS];
  __shared__ bf16 sB[2 * TILE_ELEMS];

  const int z = blockIdx.z;
  const bf16*  W    = (z == 0) ? wq : (z == 1) ? wk : wv;
  const float* bias = (z == 0) ? bq : (z == 1) ? bk : bv;
  const float  scale = (z == 0) ? 0.03125f : 1.0f;

  const int tid   = threadIdx.x;
  const int wave  = tid >> 5;
  const int lane  = tid & 31;
  const int lhalf = lane >> 4;
  const int lmod  = lane & 15;

  const int r0b = blockIdx.x * 64;
  const int c0  = blockIdx.y * 64;
  const int r0  = r0b + wave * 16;

  const v8f zf = {0.f, 0.f, 0.f, 0.f, 0.f, 0.f, 0.f, 0.f};
  v8f acc[4] = {zf, zf, zf, zf};

  gemm_tile<D_DIM>(xb + (size_t)r0b * D_DIM, D_DIM,
                   W + (size_t)c0 * D_DIM, D_DIM,
                   sA, sB, acc, tid, wave, lhalf, lmod);

  if (z == 2) {
    // Vt[b][e][s], ld = S_DIM -> P@V gets K-contiguous B fragments.
#pragma unroll
    for (int n = 0; n < 4; n++) {
      const int col = c0 + 16 * n + lmod;
      const float bb = bias[col];
#pragma unroll
      for (int j = 0; j < 8; j++) {
        const int row  = r0 + j + lhalf * 8;
        const int bidx = row >> 11;
        const int srow = row & 2047;
        vt[(size_t)bidx * D_DIM * S_DIM + (size_t)col * S_DIM + srow] =
            (bf16)(acc[n][j] + bb);
      }
    }
  } else {
    bf16* o = (z == 0) ? qo : ko;
#pragma unroll
    for (int n = 0; n < 4; n++) {
      const int col = c0 + 16 * n + lmod;
      const float bb = bias[col];
#pragma unroll
      for (int j = 0; j < 8; j++) {
        const int row = r0 + j + lhalf * 8;
        o[(size_t)row * D_DIM + col] = (bf16)((acc[n][j] + bb) * scale);
      }
    }
  }
}

// -----------------------------------------------------------------------------
// Scores + softmax: 64 q rows per block x all 2048 keys, two streaming passes
// (row max, then exp + row sum). Stores unnormalized P (bf16) and 1/rowsum.
// -----------------------------------------------------------------------------
__global__ __launch_bounds__(128) void attn_softmax(
    const bf16* __restrict__ qarr, const bf16* __restrict__ karr,
    bf16* __restrict__ parr, float* __restrict__ linv) {
  __shared__ bf16 sA[2 * TILE_ELEMS];
  __shared__ bf16 sB[2 * TILE_ELEMS];

  const int b = blockIdx.y;
  const bf16* Qb = qarr + (size_t)b * S_DIM * D_DIM;
  const bf16* Kb = karr + (size_t)b * S_DIM * D_DIM;
  bf16* Pb = parr + (size_t)b * S_DIM * S_DIM;

  const int tid   = threadIdx.x;
  const int wave  = tid >> 5;
  const int lane  = tid & 31;
  const int lhalf = lane >> 4;
  const int lmod  = lane & 15;
  const int r0b   = blockIdx.x * 64;
  const int q0    = r0b + wave * 16;

  const v8f zf = {0.f, 0.f, 0.f, 0.f, 0.f, 0.f, 0.f, 0.f};
  const bf16* Ag = Qb + (size_t)r0b * D_DIM;

  float m[8];
#pragma unroll
  for (int j = 0; j < 8; j++) m[j] = -3.0e38f;

  // Pass 1: row max
  for (int kt = 0; kt < S_DIM; kt += 64) {
    v8f s[4] = {zf, zf, zf, zf};
    gemm_tile<D_DIM>(Ag, D_DIM, Kb + (size_t)kt * D_DIM, D_DIM,
                     sA, sB, s, tid, wave, lhalf, lmod);
#pragma unroll
    for (int j = 0; j < 8; j++) {
      float v = fmaxf(fmaxf(s[0][j], s[1][j]), fmaxf(s[2][j], s[3][j]));
      m[j] = fmaxf(m[j], v);
    }
  }
#pragma unroll
  for (int j = 0; j < 8; j++) m[j] = redmax16(m[j]);

  // Pass 2: exp, row sum, store unnormalized P
  float l[8] = {0.f, 0.f, 0.f, 0.f, 0.f, 0.f, 0.f, 0.f};
  for (int kt = 0; kt < S_DIM; kt += 64) {
    v8f s[4] = {zf, zf, zf, zf};
    gemm_tile<D_DIM>(Ag, D_DIM, Kb + (size_t)kt * D_DIM, D_DIM,
                     sA, sB, s, tid, wave, lhalf, lmod);
#pragma unroll
    for (int n = 0; n < 4; n++) {
      const int col = kt + 16 * n + lmod;
#pragma unroll
      for (int j = 0; j < 8; j++) {
        float p = __expf(s[n][j] - m[j]);
        l[j] += p;
        Pb[(size_t)(q0 + j + lhalf * 8) * S_DIM + col] = (bf16)p;
      }
    }
  }
#pragma unroll
  for (int j = 0; j < 8; j++) l[j] = redsum16(l[j]);
  if (lmod == 0) {
#pragma unroll
    for (int j = 0; j < 8; j++)
      linv[b * S_DIM + q0 + j + lhalf * 8] = 1.0f / l[j];
  }
}

// -----------------------------------------------------------------------------
// O = (P @ V) * (1/l), f32 output. Vt pre-transposed -> contiguous B frags.
// -----------------------------------------------------------------------------
__global__ __launch_bounds__(128) void attn_pv(
    const bf16* __restrict__ parr, const bf16* __restrict__ vt,
    const float* __restrict__ linv, float* __restrict__ out) {
  __shared__ bf16 sA[2 * TILE_ELEMS];
  __shared__ bf16 sB[2 * TILE_ELEMS];

  const int b = blockIdx.z;
  const bf16* Pb = parr + (size_t)b * S_DIM * S_DIM;
  const bf16* Vt = vt + (size_t)b * D_DIM * S_DIM;
  float* Ob = out + (size_t)b * S_DIM * D_DIM;

  const int tid   = threadIdx.x;
  const int wave  = tid >> 5;
  const int lane  = tid & 31;
  const int lhalf = lane >> 4;
  const int lmod  = lane & 15;
  const int r0b   = blockIdx.x * 64;
  const int q0    = r0b + wave * 16;
  const int d0c   = blockIdx.y * 64;

  const v8f zf = {0.f, 0.f, 0.f, 0.f, 0.f, 0.f, 0.f, 0.f};
  v8f acc[4] = {zf, zf, zf, zf};

  gemm_tile<S_DIM>(Pb + (size_t)r0b * S_DIM, S_DIM,
                   Vt + (size_t)d0c * S_DIM, S_DIM,
                   sA, sB, acc, tid, wave, lhalf, lmod);

  float li[8];
#pragma unroll
  for (int j = 0; j < 8; j++) li[j] = linv[b * S_DIM + q0 + j + lhalf * 8];

#pragma unroll
  for (int n = 0; n < 4; n++) {
    const int col = d0c + 16 * n + lmod;
#pragma unroll
    for (int j = 0; j < 8; j++) {
      Ob[(size_t)(q0 + j + lhalf * 8) * D_DIM + col] = acc[n][j] * li[j];
    }
  }
}

// -----------------------------------------------------------------------------
extern "C" void kernel_launch(void* const* d_in, const int* in_sizes, int n_in,
                              void* d_out, int out_size, void* d_ws, size_t ws_size,
                              hipStream_t stream) {
  (void)in_sizes; (void)n_in; (void)out_size; (void)ws_size;

  const float* x  = (const float*)d_in[0];
  const float* Wq = (const float*)d_in[1];
  const float* bq = (const float*)d_in[2];
  const float* Wk = (const float*)d_in[3];
  const float* bk = (const float*)d_in[4];
  const float* Wv = (const float*)d_in[5];
  const float* bv = (const float*)d_in[6];
  float* out = (float*)d_out;

  char* ws = (char*)d_ws;
  size_t off = 0;
  bf16* xb  = (bf16*)(ws + off); off += (size_t)R_TOT * D_DIM * 2;   // 16 MB
  bf16* wqb = (bf16*)(ws + off); off += (size_t)D_DIM * D_DIM * 2;   //  2 MB
  bf16* wkb = (bf16*)(ws + off); off += (size_t)D_DIM * D_DIM * 2;
  bf16* wvb = (bf16*)(ws + off); off += (size_t)D_DIM * D_DIM * 2;
  bf16* qb  = (bf16*)(ws + off); off += (size_t)R_TOT * D_DIM * 2;   // 16 MB (pre-scaled)
  bf16* kb  = (bf16*)(ws + off); off += (size_t)R_TOT * D_DIM * 2;   // 16 MB
  bf16* vtb = (bf16*)(ws + off); off += (size_t)R_TOT * D_DIM * 2;   // 16 MB (transposed)
  bf16* pb  = (bf16*)(ws + off); off += (size_t)B_DIM * S_DIM * S_DIM * 2; // 32 MB
  float* li = (float*)(ws + off); off += (size_t)R_TOT * 4;          // 32 KB

  const int nx = R_TOT * D_DIM;
  const int nw = D_DIM * D_DIM;
  cvt_f32_bf16<<<dim3((nx + 255) / 256), dim3(256), 0, stream>>>(x, xb, nx);
  cvt_f32_bf16<<<dim3((nw + 255) / 256), dim3(256), 0, stream>>>(Wq, wqb, nw);
  cvt_f32_bf16<<<dim3((nw + 255) / 256), dim3(256), 0, stream>>>(Wk, wkb, nw);
  cvt_f32_bf16<<<dim3((nw + 255) / 256), dim3(256), 0, stream>>>(Wv, wvb, nw);

  qkv_gemm<<<dim3(R_TOT / 64, D_DIM / 64, 3), dim3(128), 0, stream>>>(
      xb, wqb, wkb, wvb, bq, bk, bv, qb, kb, vtb);

  attn_softmax<<<dim3(S_DIM / 64, B_DIM), dim3(128), 0, stream>>>(qb, kb, pb, li);

  attn_pv<<<dim3(S_DIM / 64, D_DIM / 64, B_DIM), dim3(128), 0, stream>>>(
      pb, vtb, li, out);
}
